// Generator_23235773071433
// MI455X (gfx1250) — matrix operand order repaired and checked by reading, not compiled, not run on hardware
//
#include <hip/hip_runtime.h>
#include <hip/hip_bf16.h>

// ---------------- constants ----------------
#define BB      8
#define NNODE   2000
#define NEDGE_T 32000
#define NTOT    16000            // BB*NNODE
#define BE      256000           // BB*NEDGE_T
#define NEDGES  272000           // BE + NTOT (self loops)
#define LAT     128
#define STY     64
#define NF      10
#define HEADS   4
#define HID     128
#define C1      512              // HEADS*HID
#define OUTD    13
#define KP0     96               // 74 padded to 96 (mult of 32)
#define SLOPE   0.2f

typedef _Float16 v16h __attribute__((ext_vector_type(16)));
typedef _Float16 h8   __attribute__((ext_vector_type(8)));
typedef float    v8f  __attribute__((ext_vector_type(8)));

__device__ __forceinline__ float lrelu(float x) { return x > 0.f ? x : SLOPE * x; }

// monotone float <-> uint encoding for atomic max on floats
__device__ __forceinline__ unsigned fenc(float f) {
    unsigned u = __float_as_uint(f);
    return (u >> 31) ? ~u : (u | 0x80000000u);
}
__device__ __forceinline__ float fdec(unsigned u) {
    unsigned b = (u >> 31) ? (u & 0x7FFFFFFFu) : ~u;
    return __uint_as_float(b);
}

// ---------------- utility fills ----------------
__global__ void fill_f32(float* p, float v, int n) {
    int i = blockIdx.x * blockDim.x + threadIdx.x;
    if (i < n) p[i] = v;
}
__global__ void fill_u32(unsigned* p, unsigned v, int n) {
    int i = blockIdx.x * blockDim.x + threadIdx.x;
    if (i < n) p[i] = v;
}

// ---------------- MLP: z (8x128) -> s (8x64) ----------------
__global__ void mlp_kernel(const float* __restrict__ z,
                           const float* __restrict__ w1, const float* __restrict__ b1,
                           const float* __restrict__ w2, const float* __restrict__ b2,
                           const float* __restrict__ w3, const float* __restrict__ b3,
                           float* __restrict__ s) {
    __shared__ float zr[LAT];
    __shared__ float a1[256];
    __shared__ float a2[512];
    int b = blockIdx.x, t = threadIdx.x;   // blockDim = 512
    if (t < LAT) zr[t] = z[b * LAT + t];
    __syncthreads();
    if (t < 256) {
        float acc = b1[t];
        for (int k = 0; k < LAT; ++k) acc += zr[k] * w1[k * 256 + t];
        a1[t] = lrelu(acc);
    }
    __syncthreads();
    {
        float acc = b2[t];
        for (int k = 0; k < 256; ++k) acc += a1[k] * w2[k * 512 + t];
        a2[t] = lrelu(acc);
    }
    __syncthreads();
    if (t < STY) {
        float acc = b3[t];
        for (int k = 0; k < 512; ++k) acc += a2[k] * w3[k * STY + t];
        s[b * STY + t] = acc;
    }
}

// ---------------- edge list (template edges + self loops) ----------------
__global__ void build_edges(const int* __restrict__ tei, int* __restrict__ src, int* __restrict__ dst) {
    int e = blockIdx.x * blockDim.x + threadIdx.x;
    if (e >= NEDGES) return;
    if (e < BE) {
        int b = e / NEDGE_T, j = e % NEDGE_T;
        src[e] = tei[j]           + b * NNODE;
        dst[e] = tei[NEDGE_T + j] + b * NNODE;
    } else {
        int v = e - BE;
        src[e] = v; dst[e] = v;
    }
}

// ---------------- pack H0 = [template_x | s_exp] -> f16, K padded to 96 ----------------
__global__ void pack_h0(const float* __restrict__ tx, const float* __restrict__ s,
                        _Float16* __restrict__ H0) {
    int idx = blockIdx.x * blockDim.x + threadIdx.x;
    if (idx >= NTOT * KP0) return;
    int i = idx / KP0, j = idx % KP0;
    float v = 0.f;
    if (j < NF)            v = tx[(i % NNODE) * NF + j];
    else if (j < NF + STY) v = s[(i / NNODE) * STY + (j - NF)];
    H0[idx] = (_Float16)v;
}

// ---------------- transpose-pack weight W[K x N] -> Wt[Np x Kp] f16, zero padded ----------
__global__ void pack_wt(const float* __restrict__ W, _Float16* __restrict__ Wt,
                        int K, int N, int Kp, int Np) {
    int idx = blockIdx.x * blockDim.x + threadIdx.x;
    if (idx >= Np * Kp) return;
    int n = idx / Kp, k = idx % Kp;
    Wt[idx] = (_Float16)((n < N && k < K) ? W[k * N + n] : 0.f);
}

// ---------------- WMMA GEMM: C[M x Np] = A[M x Kp] * Bt[Np x Kp]^T ----------------
// one 16x16 tile per wave, K chunks of 32, f16 in / f32 accumulate
__global__ void wmma_gemm(const _Float16* __restrict__ A, const _Float16* __restrict__ Bt,
                          float* __restrict__ C, int M, int Kp, int Np) {
    int gid  = blockIdx.x * blockDim.x + threadIdx.x;
    int wave = gid >> 5;
    int lane = threadIdx.x & 31;
    int tiles_n = Np >> 4;
    int tm = wave / tiles_n;
    int tn = wave % tiles_n;
    if (tm * 16 >= M) return;
    int hi = (lane >= 16) ? 1 : 0;
    int r  = lane & 15;
    const _Float16* arow = A  + (size_t)(tm * 16 + r) * Kp;
    const _Float16* brow = Bt + (size_t)(tn * 16 + r) * Kp;
    v8f acc = {};
    for (int kc = 0; kc < Kp; kc += 32) {
        // A 16x32 frag: half-wave 0 holds K = 0..7,16..23 ; half-wave 1 holds K = 8..15,24..31
        h8 a_lo = *(const h8*)(arow + kc + hi * 8);
        h8 a_hi = *(const h8*)(arow + kc + hi * 8 + 16);
        // B 32x16 frag: half-wave 0 holds K = 0..7,16..23 ; half-wave 1 holds K = 8..15,24..31
        h8 b_lo = *(const h8*)(brow + kc + hi * 8);
        h8 b_hi = *(const h8*)(brow + kc + hi * 8 + 16);
        v16h a, b;
#pragma unroll
        for (int i = 0; i < 8; ++i) {
            a[i] = a_lo[i]; a[i + 8] = a_hi[i];
            b[i] = b_lo[i]; b[i + 8] = b_hi[i];
        }
        acc = __builtin_amdgcn_wmma_f32_16x16x32_f16(
            /*neg_a=*/false, a, /*neg_b=*/false, b,
            /*c_mod=*/(short)0, acc, /*reuse_a=*/false, /*reuse_b=*/false);
    }
    int n = tn * 16 + r;
#pragma unroll
    for (int i = 0; i < 8; ++i) {
        int m = tm * 16 + i + hi * 8;
        C[(size_t)m * Np + n] = acc[i];
    }
}

// ---------------- attention coefficients per (node, head) ----------------
__global__ void attn_coef(const float* __restrict__ h, const float* __restrict__ a_s,
                          const float* __restrict__ a_d, float* __restrict__ als,
                          float* __restrict__ ald, int heads, int hid, int stride) {
    int idx = blockIdx.x * blockDim.x + threadIdx.x;
    if (idx >= NTOT * heads) return;
    int i = idx / heads, hh = idx % heads;
    const float* hp  = h   + (size_t)i * stride + hh * hid;
    const float* asp = a_s + hh * hid;
    const float* adp = a_d + hh * hid;
    float s0 = 0.f, s1 = 0.f;
    for (int c = 0; c < hid; ++c) { float v = hp[c]; s0 += v * asp[c]; s1 += v * adp[c]; }
    als[idx] = s0; ald[idx] = s1;
}

// ---------------- edge softmax phase ----------------
__global__ void edge_logit_max(const float* __restrict__ als, const float* __restrict__ ald,
                               const int* __restrict__ src, const int* __restrict__ dst,
                               float* __restrict__ logits, unsigned* __restrict__ mx, int heads) {
    int idx = blockIdx.x * blockDim.x + threadIdx.x;
    if (idx >= NEDGES * heads) return;
    int e = idx / heads, hh = idx % heads;
    float lg = lrelu(als[src[e] * heads + hh] + ald[dst[e] * heads + hh]);
    logits[idx] = lg;
    atomicMax(&mx[dst[e] * heads + hh], fenc(lg));
}

__global__ void edge_alpha(float* __restrict__ logits, const unsigned* __restrict__ mx,
                           float* __restrict__ den, const int* __restrict__ dst, int heads) {
    int idx = blockIdx.x * blockDim.x + threadIdx.x;
    if (idx >= NEDGES * heads) return;
    int e = idx / heads, hh = idx % heads;
    float a = expf(logits[idx] - fdec(mx[dst[e] * heads + hh]));
    logits[idx] = a;
    atomicAdd(&den[dst[e] * heads + hh], a);
}

__global__ void edge_norm(float* __restrict__ alpha, const float* __restrict__ den,
                          const int* __restrict__ dst, int heads) {
    int idx = blockIdx.x * blockDim.x + threadIdx.x;
    if (idx >= NEDGES * heads) return;
    int e = idx / heads, hh = idx % heads;
    alpha[idx] /= den[dst[e] * heads + hh];
}

// ---------------- weighted message scatter (channel parallel, atomics in L2) ----------
__global__ void scatter_msgs(const float* __restrict__ alpha, const float* __restrict__ h,
                             float* __restrict__ agg, const int* __restrict__ src,
                             const int* __restrict__ dst, int C, int hid, int heads, int stride) {
    int idx = blockIdx.x * blockDim.x + threadIdx.x;
    if (idx >= NEDGES * C) return;
    int e = idx / C, c = idx % C;
    int hh = c / hid;
    atomicAdd(&agg[(size_t)dst[e] * stride + c],
              alpha[e * heads + hh] * h[(size_t)src[e] * stride + c]);
}

// ---------------- bias + leaky + (residual) + layernorm, emit f32 + f16 ----------------
__global__ void finalize_ln(const float* __restrict__ agg, const float* __restrict__ bias,
                            const float* __restrict__ gam, const float* __restrict__ bet,
                            const float* __restrict__ resin, int use_res,
                            float* __restrict__ out, _Float16* __restrict__ outh) {
    __shared__ float red[256];
    int i = blockIdx.x, t = threadIdx.x;   // blockDim = 256, C1 = 512
    float v0 = lrelu(agg[(size_t)i * C1 + t]       + bias[t]);
    float v1 = lrelu(agg[(size_t)i * C1 + t + 256] + bias[t + 256]);
    if (use_res) {
        v0 += resin[(size_t)i * C1 + t];
        v1 += resin[(size_t)i * C1 + t + 256];
    }
    red[t] = v0 + v1; __syncthreads();
    for (int s = 128; s > 0; s >>= 1) { if (t < s) red[t] += red[t + s]; __syncthreads(); }
    float mu = red[0] * (1.f / C1); __syncthreads();
    float d0 = v0 - mu, d1 = v1 - mu;
    red[t] = d0 * d0 + d1 * d1; __syncthreads();
    for (int s = 128; s > 0; s >>= 1) { if (t < s) red[t] += red[t + s]; __syncthreads(); }
    float rs = rsqrtf(red[0] * (1.f / C1) + 1e-5f);
    float y0 = d0 * rs * gam[t]       + bet[t];
    float y1 = d1 * rs * gam[t + 256] + bet[t + 256];
    out[(size_t)i * C1 + t]        = y0;
    out[(size_t)i * C1 + t + 256]  = y1;
    outh[(size_t)i * C1 + t]       = (_Float16)y0;
    outh[(size_t)i * C1 + t + 256] = (_Float16)y1;
}

// ---------------- final outputs: X_gen, p_gen ----------------
__global__ void final_out(const float* __restrict__ agg3, const float* __restrict__ b3,
                          const float* __restrict__ tx, float* __restrict__ Xg,
                          float* __restrict__ pg) {
    int i = blockIdx.x * blockDim.x + threadIdx.x;
    if (i >= NTOT) return;
    int ti = i % NNODE;
    float H[OUTD];
#pragma unroll
    for (int c = 0; c < OUTD; ++c) H[c] = agg3[i * 16 + c] + b3[c];
    float p0 = tx[ti * NF + 0] + H[0];
    float p1 = tx[ti * NF + 1] + H[1];
    float p2 = tx[ti * NF + 2] + H[2];
    pg[i * 3 + 0] = p0; pg[i * 3 + 1] = p1; pg[i * 3 + 2] = p2;
    Xg[i * NF + 0] = p0; Xg[i * NF + 1] = p1; Xg[i * NF + 2] = p2;
#pragma unroll
    for (int j = 0; j < 7; ++j)
        Xg[i * NF + 3 + j] = tx[ti * NF + 3 + j] + H[6 + j];  // delta_f[:,3:] = H_out[:,6:13]
}

// ---------------- EF = p_gen[dst] - p_gen[src] over template edges ----------------
__global__ void edge_ef(const float* __restrict__ pg, const int* __restrict__ src,
                        const int* __restrict__ dst, float* __restrict__ EF) {
    int e = blockIdx.x * blockDim.x + threadIdx.x;
    if (e >= BE) return;
    int s = src[e], d = dst[e];
#pragma unroll
    for (int k = 0; k < 3; ++k) EF[e * 3 + k] = pg[d * 3 + k] - pg[s * 3 + k];
}

// ================= host side =================
static inline int cdiv(long long a, int b) { return (int)((a + b - 1) / b); }

static void run_gemm(const _Float16* A, const _Float16* Bt, float* C,
                     int M, int Kp, int Np, hipStream_t stream) {
    int waves  = (M / 16) * (Np / 16);
    int blocks = (waves + 7) / 8;
    wmma_gemm<<<blocks, 256, 0, stream>>>(A, Bt, C, M, Kp, Np);
}

static void run_gat_edges(const float* h, int stride, int C, int heads, int hid,
                          const float* a_s, const float* a_d, float* agg, int aggN,
                          float* als, float* ald, unsigned* mxb, float* den, float* alphab,
                          const int* srci, const int* dsti, hipStream_t stream) {
    fill_f32<<<cdiv(aggN, 256), 256, 0, stream>>>(agg, 0.f, aggN);
    fill_u32<<<cdiv(NTOT * heads, 256), 256, 0, stream>>>(mxb, 0x007FFFFFu, NTOT * heads); // enc(-inf)
    fill_f32<<<cdiv(NTOT * heads, 256), 256, 0, stream>>>(den, 0.f, NTOT * heads);
    attn_coef<<<cdiv(NTOT * heads, 256), 256, 0, stream>>>(h, a_s, a_d, als, ald, heads, hid, stride);
    edge_logit_max<<<cdiv((long long)NEDGES * heads, 256), 256, 0, stream>>>(als, ald, srci, dsti, alphab, mxb, heads);
    edge_alpha<<<cdiv((long long)NEDGES * heads, 256), 256, 0, stream>>>(alphab, mxb, den, dsti, heads);
    edge_norm<<<cdiv((long long)NEDGES * heads, 256), 256, 0, stream>>>(alphab, den, dsti, heads);
    scatter_msgs<<<cdiv((long long)NEDGES * C, 256), 256, 0, stream>>>(alphab, h, agg, srci, dsti, C, hid, heads, stride);
}

extern "C" void kernel_launch(void* const* d_in, const int* in_sizes, int n_in,
                              void* d_out, int out_size, void* d_ws, size_t ws_size,
                              hipStream_t stream) {
    const float* z    = (const float*)d_in[0];
    const float* tx   = (const float*)d_in[1];
    const int*   tei  = (const int*)  d_in[2];
    const float* w1   = (const float*)d_in[3];
    const float* b1   = (const float*)d_in[4];
    const float* w2   = (const float*)d_in[5];
    const float* b2   = (const float*)d_in[6];
    const float* w3   = (const float*)d_in[7];
    const float* b3   = (const float*)d_in[8];
    const float* g1w  = (const float*)d_in[9];
    const float* g1as = (const float*)d_in[10];
    const float* g1ad = (const float*)d_in[11];
    const float* g1b  = (const float*)d_in[12];
    const float* ln1g = (const float*)d_in[13];
    const float* ln1b = (const float*)d_in[14];
    const float* g2w  = (const float*)d_in[15];
    const float* g2as = (const float*)d_in[16];
    const float* g2ad = (const float*)d_in[17];
    const float* g2b  = (const float*)d_in[18];
    const float* ln2g = (const float*)d_in[19];
    const float* ln2b = (const float*)d_in[20];
    const float* g3w  = (const float*)d_in[21];
    const float* g3as = (const float*)d_in[22];
    const float* g3ad = (const float*)d_in[23];
    const float* g3b  = (const float*)d_in[24];

    // workspace carve
    char* p = (char*)d_ws;
    auto alloc = [&](size_t bytes) -> void* {
        void* r = (void*)p;
        p += (bytes + 255) & ~(size_t)255;
        return r;
    };
    int*       srci   = (int*)      alloc((size_t)NEDGES * 4);
    int*       dsti   = (int*)      alloc((size_t)NEDGES * 4);
    float*     sbuf   = (float*)    alloc((size_t)BB * STY * 4);
    _Float16*  H0h    = (_Float16*) alloc((size_t)NTOT * KP0 * 2);
    _Float16*  W1t    = (_Float16*) alloc((size_t)C1 * KP0 * 2);
    _Float16*  W2t    = (_Float16*) alloc((size_t)C1 * C1 * 2);
    _Float16*  W3t    = (_Float16*) alloc((size_t)16 * C1 * 2);
    float*     hbuf   = (float*)    alloc((size_t)NTOT * C1 * 4);
    float*     aggbuf = (float*)    alloc((size_t)NTOT * C1 * 4);
    float*     resbuf = (float*)    alloc((size_t)NTOT * C1 * 4);
    _Float16*  acth   = (_Float16*) alloc((size_t)NTOT * C1 * 2);
    float*     als    = (float*)    alloc((size_t)NTOT * HEADS * 4);
    float*     ald    = (float*)    alloc((size_t)NTOT * HEADS * 4);
    unsigned*  mxb    = (unsigned*) alloc((size_t)NTOT * HEADS * 4);
    float*     den    = (float*)    alloc((size_t)NTOT * HEADS * 4);
    float*     alphab = (float*)    alloc((size_t)NEDGES * HEADS * 4);
    float*     h3     = (float*)    alloc((size_t)NTOT * 16 * 4);
    float*     agg3   = (float*)    alloc((size_t)NTOT * 16 * 4);

    float* Xg = (float*)d_out;          // 16000 x 10
    float* pg = Xg + NTOT * NF;         // 16000 x 3
    float* EF = pg + NTOT * 3;          // 256000 x 3

    // 1) MLP + edges + packing
    mlp_kernel<<<BB, 512, 0, stream>>>(z, w1, b1, w2, b2, w3, b3, sbuf);
    build_edges<<<cdiv(NEDGES, 256), 256, 0, stream>>>(tei, srci, dsti);
    pack_h0<<<cdiv((long long)NTOT * KP0, 256), 256, 0, stream>>>(tx, sbuf, H0h);
    pack_wt<<<cdiv((long long)C1 * KP0, 256), 256, 0, stream>>>(g1w, W1t, NF + STY, C1, KP0, C1);
    pack_wt<<<cdiv((long long)C1 * C1, 256), 256, 0, stream>>>(g2w, W2t, C1, C1, C1, C1);
    pack_wt<<<cdiv((long long)16 * C1, 256), 256, 0, stream>>>(g3w, W3t, C1, OUTD, C1, 16);

    // 2) GAT layer 1
    run_gemm(H0h, W1t, hbuf, NTOT, KP0, C1, stream);
    run_gat_edges(hbuf, C1, C1, HEADS, HID, g1as, g1ad, aggbuf, NTOT * C1,
                  als, ald, mxb, den, alphab, srci, dsti, stream);
    finalize_ln<<<NTOT, 256, 0, stream>>>(aggbuf, g1b, ln1g, ln1b, nullptr, 0, resbuf, acth);

    // 3) GAT layer 2 (+ residual with H1n)
    run_gemm(acth, W2t, hbuf, NTOT, C1, C1, stream);
    run_gat_edges(hbuf, C1, C1, HEADS, HID, g2as, g2ad, aggbuf, NTOT * C1,
                  als, ald, mxb, den, alphab, srci, dsti, stream);
    finalize_ln<<<NTOT, 256, 0, stream>>>(aggbuf, g2b, ln2g, ln2b, resbuf, 1, resbuf, acth);

    // 4) GAT layer 3 (heads=1, out=13 padded to 16)
    run_gemm(acth, W3t, h3, NTOT, C1, 16, stream);
    run_gat_edges(h3, 16, OUTD, 1, OUTD, g3as, g3ad, agg3, NTOT * 16,
                  als, ald, mxb, den, alphab, srci, dsti, stream);

    // 5) outputs
    final_out<<<cdiv(NTOT, 256), 256, 0, stream>>>(agg3, g3b, tx, Xg, pg);
    edge_ef<<<cdiv(BE, 256), 256, 0, stream>>>(pg, srci, dsti, EF);
}
